// SAGE_4879082848348
// MI455X (gfx1250) — matrix operand order repaired and checked by reading, not compiled, not run on hardware
//
#include <hip/hip_runtime.h>
#include <stdint.h>

// ---------------------------------------------------------------------------
// GraphSAGE 3-layer forward for MI455X (gfx1250, wave32, WMMA).
// per-layer:  out = [h_dst | mean(h_neigh)] @ [[W_self],[W_neigh]] + b
// -> one K-doubled GEMM per layer on v_wmma_f32_16x16x32_bf16 (f32 accum),
// bf16 intermediates, async global->LDS tile copies with double buffering.
// ---------------------------------------------------------------------------

#define FANOUT 10
#define N0 1362944
#define N1 123904
#define N2 11264
#define SEEDS 1024
#define IN_F 256
#define H_F 512
#define C_F 256

typedef __attribute__((ext_vector_type(16))) __bf16 v16bf;
typedef __attribute__((ext_vector_type(8)))  float  v8f;
typedef __attribute__((ext_vector_type(4)))  int    v4i;

union FragBF { v16bf v; uint4 q[2]; };

#define AS1 __attribute__((address_space(1)))
#define AS3 __attribute__((address_space(3)))

__device__ __forceinline__ unsigned short f2bf(float f) {
    union { float f; uint32_t u; } v; v.f = f;
    uint32_t u = v.u;
    return (unsigned short)((u + 0x7FFFu + ((u >> 16) & 1u)) >> 16);
}
__device__ __forceinline__ float bf2f(unsigned short h) {
    union { uint32_t u; float f; } v; v.u = ((uint32_t)h) << 16;
    return v.f;
}

// ---- async global->LDS 16B copy (CDNA5 GLOBAL_LOAD_ASYNC_TO_LDS_B128) ----
__device__ __forceinline__ void async_copy16(const void* g, void* l) {
#if defined(__has_builtin) && __has_builtin(__builtin_amdgcn_global_load_async_to_lds_b128)
    // signature (probe-confirmed): (v4i AS1* src, v4i AS3* dst, imm off, imm cpol)
    AS1 v4i* gp = (AS1 v4i*)(uintptr_t)g;              // global flat == AS1 value
    AS3 v4i* lp = (AS3 v4i*)(uint32_t)(uintptr_t)l;    // low 32b = LDS offset
    __builtin_amdgcn_global_load_async_to_lds_b128(gp, lp, 0, 0);
#else
    uint64_t ga = (uint64_t)(uintptr_t)g;
    uint32_t la = (uint32_t)(uintptr_t)l;
    asm volatile("global_load_async_to_lds_b128 %0, %1, off"
                 :: "v"(la), "v"(ga) : "memory");
#endif
}
__device__ __forceinline__ void wait_async0() {
#if defined(__has_builtin) && __has_builtin(__builtin_amdgcn_s_wait_asynccnt)
    __builtin_amdgcn_s_wait_asynccnt(0);
#else
    asm volatile("s_wait_asynccnt 0" ::: "memory");
#endif
}

// ---------------------------------------------------------------------------
// Fold W_self (K1 x N) and W_neigh (K1 x N), both f32 row-major, into a
// transposed bf16 B matrix Bt[N][2*K1]  (each WMMA lane then reads a
// contiguous K-run, matching the CDNA5 16-bit B-fragment layout).
// ---------------------------------------------------------------------------
__global__ __launch_bounds__(256) void build_bt_kernel(
    const float* __restrict__ Wself, const float* __restrict__ Wneigh,
    unsigned short* __restrict__ Bt, int K1, int N)
{
    int K = 2 * K1;
    size_t total = (size_t)K * (size_t)N;
    size_t idx = (size_t)blockIdx.x * blockDim.x + threadIdx.x;
    if (idx >= total) return;
    int n = (int)(idx / K);
    int k = (int)(idx % K);
    float v = (k < K1) ? Wself[(size_t)k * N + n]
                       : Wneigh[(size_t)(k - K1) * N + n];
    Bt[idx] = f2bf(v);
}

// ---------------------------------------------------------------------------
// Build packed bf16 A matrix for one layer:
//   A[m][0 .. K1)     = src[m][:]                 (self features)
//   A[m][K1 .. 2*K1)  = mean_f src[edge[m,f]][:]  (neighbor mean)
// One block per destination node; coalesced along the feature dim; edge
// indices are blockIdx-uniform -> scalar loads.
// ---------------------------------------------------------------------------
template <bool SRC_F32>
__global__ __launch_bounds__(256) void build_a_kernel(
    const void* __restrict__ src_, const int* __restrict__ edges,
    unsigned short* __restrict__ A, int K1)
{
    const float*          sf = (const float*)src_;
    const unsigned short* sb = (const unsigned short*)src_;
    int m  = blockIdx.x;
    int K2 = 2 * K1;

    int e[FANOUT];
#pragma unroll
    for (int f = 0; f < FANOUT; ++f) e[f] = edges[m * FANOUT + f];

    for (int c = threadIdx.x; c < K1; c += blockDim.x) {
        float self = SRC_F32 ? sf[(size_t)m * K1 + c]
                             : bf2f(sb[(size_t)m * K1 + c]);
        float s = 0.0f;
#pragma unroll
        for (int f = 0; f < FANOUT; ++f) {
            size_t off = (size_t)e[f] * K1 + c;
            s += SRC_F32 ? sf[off] : bf2f(sb[off]);
        }
        A[(size_t)m * K2 + c]      = f2bf(self);
        A[(size_t)m * K2 + K1 + c] = f2bf(s * (1.0f / FANOUT));
    }
}

// ---------------------------------------------------------------------------
// bf16 WMMA GEMM:  out[M][N] = act( A[M][K] @ Bt[N][K]^T + bias[N] )
// Block = 256 threads = 8 wave32s laid out 4(rows) x 2(cols).
// Tile: BM=128, BN=64, BK=64. Each wave -> 32x32 patch = 2x2 WMMA tiles.
// Double-buffered LDS fed by async global->LDS b128 copies (ASYNCcnt).
// M%128==0, N%64==0, K%64==0 for all three layers -> EXEC stays all-ones.
// ---------------------------------------------------------------------------
template <bool RELU, bool OUT_BF16>
__global__ __launch_bounds__(256) void gemm_wmma_kernel(
    const unsigned short* __restrict__ A,   // [M][K] bf16 row-major
    const unsigned short* __restrict__ Bt,  // [N][K] bf16 row-major (B^T)
    const float* __restrict__ bias,         // [N]
    void* __restrict__ out_,                // [M][N] bf16 or f32
    int M, int N, int K)
{
    constexpr int BM = 128, BN = 64, BK = 64;
    constexpr int LDA = BK + 8;   // bf16 elems; pad dodges bank conflicts
    constexpr int LDB = BK + 8;

    __shared__ unsigned short sA[2][BM * LDA];   // 2 * 18.0 KB
    __shared__ unsigned short sB[2][BN * LDB];   // 2 *  9.0 KB

    const int t    = threadIdx.x;
    const int lane = t & 31;
    const int wid  = t >> 5;
    const int wm   = wid & 3;    // wave row 0..3  (32 rows each)
    const int wn   = wid >> 2;   // wave col 0..1  (32 cols each)

    const int row0 = blockIdx.x * BM;
    const int col0 = blockIdx.y * BN;

    const int halfSel = lane >> 4;   // 0: lanes 0-15, 1: lanes 16-31
    const int laneLo  = lane & 15;

    v8f acc[2][2] = {};

    auto prefetch_tile = [&](int kk0, int buf) {
        // A tile: 128 rows x 64 bf16 = 1024 x 16B chunks (4 per thread)
#pragma unroll
        for (int i = 0; i < 4; ++i) {
            int c = t + 256 * i;
            int r = c >> 3;
            int cc = (c & 7) * 8;
            async_copy16(&A[(size_t)(row0 + r) * K + kk0 + cc],
                         &sA[buf][r * LDA + cc]);
        }
        // B^T tile: 64 rows x 64 bf16 = 512 x 16B chunks (2 per thread)
#pragma unroll
        for (int i = 0; i < 2; ++i) {
            int c = t + 256 * i;
            int r = c >> 3;
            int cc = (c & 7) * 8;
            async_copy16(&Bt[(size_t)(col0 + r) * K + kk0 + cc],
                         &sB[buf][r * LDB + cc]);
        }
    };

    const int nt = K / BK;
    prefetch_tile(0, 0);

    for (int it = 0; it < nt; ++it) {
        const int cur = it & 1;
        // this wave's async writes into sA/sB[cur] are done ...
        wait_async0();
        // ... and published to all waves; also: every wave is done reading
        // buffer cur^1 (its frag ds_loads were waited before barrier-signal)
        __syncthreads();
        if (it + 1 < nt) prefetch_tile((it + 1) * BK, cur ^ 1);

#pragma unroll
        for (int ks = 0; ks < BK; ks += 32) {
            // B fragments (lane holds N = lane&15; K chunks at
            // 8*(lane>>4) and 16 + 8*(lane>>4))
            FragBF bfr[2];
#pragma unroll
            for (int tj = 0; tj < 2; ++tj) {
                int bn = wn * 32 + tj * 16 + laneLo;
                int bk = ks + (halfSel << 3);
                bfr[tj].q[0] = *(const uint4*)&sB[cur][bn * LDB + bk];
                bfr[tj].q[1] = *(const uint4*)&sB[cur][bn * LDB + bk + 16];
            }
#pragma unroll
            for (int ti = 0; ti < 2; ++ti) {
                // A fragment (lane holds M = lane&15; K chunks at
                // 8*(lane>>4) and 16 + 8*(lane>>4))
                FragBF af;
                int ar = wm * 32 + ti * 16 + laneLo;
                int ak = ks + (halfSel << 3);
                af.q[0] = *(const uint4*)&sA[cur][ar * LDA + ak];
                af.q[1] = *(const uint4*)&sA[cur][ar * LDA + ak + 16];
#pragma unroll
                for (int tj = 0; tj < 2; ++tj) {
                    acc[ti][tj] = __builtin_amdgcn_wmma_f32_16x16x32_bf16(
                        false, af.v, false, bfr[tj].v,
                        (short)0, acc[ti][tj], false, false);
                }
            }
        }
    }

    // ---- epilogue: C/D layout -> col = lane&15, rows = 8*(lane>>4)+r ----
#pragma unroll
    for (int ti = 0; ti < 2; ++ti) {
#pragma unroll
        for (int tj = 0; tj < 2; ++tj) {
            int gm0 = row0 + wm * 32 + ti * 16 + (halfSel << 3);
            int gn  = col0 + wn * 32 + tj * 16 + laneLo;
            float bv = bias[gn];
            v8f a = acc[ti][tj];
#pragma unroll
            for (int r = 0; r < 8; ++r) {
                float v = a[r] + bv;
                if (RELU) v = fmaxf(v, 0.0f);
                size_t o = (size_t)(gm0 + r) * N + gn;
                if (OUT_BF16) ((unsigned short*)out_)[o] = f2bf(v);
                else          ((float*)out_)[o] = v;
            }
        }
    }
}

// ---------------------------------------------------------------------------
// Host-side orchestration (graph-capture safe: only pointer math + launches)
// ---------------------------------------------------------------------------
extern "C" void kernel_launch(void* const* d_in, const int* in_sizes, int n_in,
                              void* d_out, int out_size, void* d_ws, size_t ws_size,
                              hipStream_t stream)
{
    (void)in_sizes; (void)n_in; (void)out_size; (void)ws_size;

    const float* x   = (const float*)d_in[0];
    const float* Ws0 = (const float*)d_in[1];
    const float* Wn0 = (const float*)d_in[2];
    const float* b0  = (const float*)d_in[3];
    const float* Ws1 = (const float*)d_in[4];
    const float* Wn1 = (const float*)d_in[5];
    const float* b1  = (const float*)d_in[6];
    const float* Ws2 = (const float*)d_in[7];
    const float* Wn2 = (const float*)d_in[8];
    const float* b2  = (const float*)d_in[9];
    const int*   e0  = (const int*)d_in[10];
    const int*   e1  = (const int*)d_in[11];
    const int*   e2  = (const int*)d_in[12];
    float* out = (float*)d_out;

    // ---- workspace layout (256-byte aligned slabs) ----
    char* ws = (char*)d_ws;
    size_t off = 0;
    auto take = [&](size_t bytes) -> char* {
        char* p = ws + off;
        off = (off + bytes + 255) & ~(size_t)255;
        return p;
    };
    unsigned short* Bt0 = (unsigned short*)take((size_t)H_F * (2 * IN_F) * 2); // 512x512
    unsigned short* Bt1 = (unsigned short*)take((size_t)H_F * (2 * H_F) * 2);  // 512x1024
    unsigned short* Bt2 = (unsigned short*)take((size_t)C_F * (2 * H_F) * 2);  // 256x1024
    unsigned short* A0  = (unsigned short*)take((size_t)N1 * (2 * IN_F) * 2);  // 123904x512
    unsigned short* h1  = (unsigned short*)take((size_t)N1 * H_F * 2);         // 123904x512
    unsigned short* A1  = (unsigned short*)take((size_t)N2 * (2 * H_F) * 2);   // 11264x1024
    unsigned short* h2  = (unsigned short*)take((size_t)N2 * H_F * 2);         // 11264x512
    unsigned short* A2  = (unsigned short*)take((size_t)SEEDS * (2 * H_F) * 2);// 1024x1024

    // ---- weight folding / transpose / bf16 conversion ----
    {
        size_t n0 = (size_t)(2 * IN_F) * H_F;
        build_bt_kernel<<<(unsigned)((n0 + 255) / 256), 256, 0, stream>>>(Ws0, Wn0, Bt0, IN_F, H_F);
        size_t n1e = (size_t)(2 * H_F) * H_F;
        build_bt_kernel<<<(unsigned)((n1e + 255) / 256), 256, 0, stream>>>(Ws1, Wn1, Bt1, H_F, H_F);
        size_t n2e = (size_t)(2 * H_F) * C_F;
        build_bt_kernel<<<(unsigned)((n2e + 255) / 256), 256, 0, stream>>>(Ws2, Wn2, Bt2, H_F, C_F);
    }

    // ---- layer 0 ----
    build_a_kernel<true><<<N1, 256, 0, stream>>>(x, e0, A0, IN_F);
    gemm_wmma_kernel<true, true>
        <<<dim3(N1 / 128, H_F / 64), 256, 0, stream>>>(A0, Bt0, b0, h1, N1, H_F, 2 * IN_F);

    // ---- layer 1 ----
    build_a_kernel<false><<<N2, 256, 0, stream>>>(h1, e1, A1, H_F);
    gemm_wmma_kernel<true, true>
        <<<dim3(N2 / 128, H_F / 64), 256, 0, stream>>>(A1, Bt1, b1, h2, N2, H_F, 2 * H_F);

    // ---- layer 2 (f32 output, no activation) ----
    build_a_kernel<false><<<SEEDS, 256, 0, stream>>>(h2, e2, A2, H_F);
    gemm_wmma_kernel<false, false>
        <<<dim3(SEEDS / 128, C_F / 64), 256, 0, stream>>>(A2, Bt2, b2, out, SEEDS, C_F, 2 * H_F);
}
